// DGCLCMIWithGatedMechanism_67276367724974
// MI455X (gfx1250) — matrix operand order repaired and checked by reading, not compiled, not run on hardware
//
#include <hip/hip_runtime.h>
#include <hip/hip_bf16.h>
#include <stddef.h>

// ---------------------------------------------------------------------------
// Model constants (from reference)
// ---------------------------------------------------------------------------
#define N_USER   5000
#define N_ITEM   15000
#define NN       (N_USER + N_ITEM)   // 20000
#define EMB      128
#define HIDD     512
#define STRUCT_D 256
#define SEQL     8
#define FEAT     64
#define NHEADS   4
#define HEADD    128                  // HID / NHEAD
#define NNZ_E    640000
#define BATCH_B  4096
#define MAXTOK   (N_ITEM * SEQL)      // 120000

// ---------------------------------------------------------------------------
// Input pytree flattening (top-level insertion order; dicts sorted by key,
// ASCII so 'W_*' < 'b_*'; lists in index order).
// ---------------------------------------------------------------------------
enum {
  IDX_CIRC_SEQ = 0, IDX_MIR_SEQ = 1, IDX_CIRC_STRUCT = 2, IDX_MIR_STRUCT = 3,
  IDX_CIRC_TF = 4,            // 45 leaves
  IDX_MIR_TF  = 49,           // 45 leaves
  IDX_CIRC_FUS = 94,          // 14 leaves
  IDX_MIR_FUS  = 108,         // 14 leaves
  IDX_PROP     = 122,         // 3 * 27 leaves
  IDX_ADJ_VALS = 203, IDX_ADJ_ROW = 204, IDX_ADJ_COL = 205,
  IDX_USERS = 206, IDX_ITEMS = 207
};
// transformer params, sorted-key order
enum {
  TF_FC_MID_B = 0, TF_FC_MID_W = 1, TF_FC_OUT_B = 2, TF_FC_OUT_W = 3,
  TF_IN_LN_B = 4, TF_IN_LN_G = 5, TF_LAYER0 = 6, TF_LAYER1 = 22,
  TF_LN_B = 38, TF_LN_G = 39, TF_OUT_LN_B = 40, TF_OUT_LN_G = 41,
  TF_POOL_W = 42, TF_PROJ_B = 43, TF_PROJ_W = 44
};
enum {
  L_BK = 0, L_BO = 1, L_BQ = 2, L_BV = 3, L_FF1_B = 4, L_FF1_W = 5,
  L_FF2_B = 6, L_FF2_W = 7, L_LN1_B = 8, L_LN1_G = 9, L_LN2_B = 10,
  L_LN2_G = 11, L_WK = 12, L_WO = 13, L_WQ = 14, L_WV = 15
};
enum {
  FU_F_B = 0, FU_F_LN_B = 1, FU_F_LN_G = 2, FU_F_W = 3, FU_S_B = 4,
  FU_S_LN_B = 5, FU_S_LN_G = 6, FU_S_W = 7, FU_SW = 8, FU_T_B = 9,
  FU_T_LN_B = 10, FU_T_LN_G = 11, FU_T_W = 12, FU_TW = 13
};
enum {
  P_W_C = 0, P_W_C_TO_M = 1, P_W_EC_C = 2, P_W_EC_M = 3, P_W_F_C = 4,
  P_W_F_M = 5, P_W_I_C = 6, P_W_I_M = 7, P_W_M = 8, P_W_M_TO_C = 9,
  P_W_O_C = 10, P_W_O_M = 11, P_B_C = 12, P_B_C_TO_M = 13, P_B_EC_C = 14,
  P_B_EC_M = 15, P_B_F_C = 16, P_B_F_M = 17, P_B_I_C = 18, P_B_I_M = 19,
  P_B_M = 20, P_B_M_TO_C = 21, P_B_O_C = 22, P_B_O_M = 23, P_LN_B = 24,
  P_LN_G = 25, P_SCALE = 26
};

// ---------------------------------------------------------------------------
// WMMA GEMM: C[M,N] = act(A[M,K] @ B[K,N] + bias)
// Requirements: K % 32 == 0, N % 64 == 0 (true for all GEMMs here).
// fp32 in/out, f16 compute via v_wmma_f32_16x16x32_f16, f32 accumulate.
// 256 threads = 8 waves (4x2). Block tile 128x64; wave tile 32x32
// (2x2 WMMA 16x16 tiles -> 4 WMMAs / wave / K-step, fragments reused 2x).
// Double-buffered LDS -> one barrier per K-step; next tile's global loads
// are issued before the WMMA block to overlap VMEM latency.
// ---------------------------------------------------------------------------
typedef __attribute__((ext_vector_type(16))) _Float16 v16h;
typedef __attribute__((ext_vector_type(8)))  float    v8f;

union FragH { v16h v; unsigned int u[8]; };
union AccF  { v8f  v; float f[8]; };

static __device__ __forceinline__ unsigned short f2h(float x) {
  union { _Float16 h; unsigned short u; } c; c.h = (_Float16)x; return c.u;
}

#define GT_M 128
#define GT_N 64
#define GT_K 32
#define A_STR 40   // LDS row stride (f16 elems); even; rows stay 16B aligned
#define B_STR 40

template <int ACT>   // 0 none, 1 relu, 2 tanh
__global__ __launch_bounds__(256) void gemm_wmma_f16(
    const float* __restrict__ A, const float* __restrict__ B,
    const float* __restrict__ bias, float* __restrict__ C,
    int M, int K, int N)
{
  __shared__ alignas(16) unsigned short lA[2][GT_M * A_STR];  // 2 x 10240 B
  __shared__ alignas(16) unsigned short lB[2][GT_N * B_STR];  // 2 x  5120 B

  const int m0 = blockIdx.y * GT_M;
  const int n0 = blockIdx.x * GT_N;
  const int t    = threadIdx.x;
  const int lane = t & 31;
  const int w    = t >> 5;
  const int wm   = (w & 3) * 32;   // 4 waves along M
  const int wn   = (w >> 2) * 32;  // 2 waves along N
  const int fm   = lane & 15;
  const int fh   = lane >> 4;

  // cooperative-load coordinates
  const int ar = t & 127;          // A row within tile
  const int ac = (t >> 7) * 16;    // A col start (16 floats per thread)
  const int bk = t >> 3;           // B k-row 0..31
  const int bn = (t & 7) * 8;      // B col start (8 floats per thread)

  float a_reg[16];
  float b_reg[8];

  // ---- global load of tile kt into regs ----
  auto load_tile = [&](int k0) {
    const int gr = m0 + ar;
    if (gr < M) {
      const float4* s4 = (const float4*)(A + (size_t)gr * K + k0 + ac);
#pragma unroll
      for (int i = 0; i < 4; ++i) {
        float4 p = s4[i];
        a_reg[4 * i + 0] = p.x; a_reg[4 * i + 1] = p.y;
        a_reg[4 * i + 2] = p.z; a_reg[4 * i + 3] = p.w;
      }
    } else {
#pragma unroll
      for (int i = 0; i < 16; ++i) a_reg[i] = 0.0f;
    }
    const float4* s4 = (const float4*)(B + (size_t)(k0 + bk) * N + n0 + bn);
#pragma unroll
    for (int i = 0; i < 2; ++i) {
      float4 p = s4[i];
      b_reg[4 * i + 0] = p.x; b_reg[4 * i + 1] = p.y;
      b_reg[4 * i + 2] = p.z; b_reg[4 * i + 3] = p.w;
    }
  };
  // ---- convert regs -> f16 and stage into LDS buffer `buf` ----
  auto store_tile = [&](int buf) {
    unsigned int* d32 = (unsigned int*)&lA[buf][ar * A_STR + ac];
#pragma unroll
    for (int i = 0; i < 8; ++i)
      d32[i] = (unsigned)f2h(a_reg[2 * i]) | ((unsigned)f2h(a_reg[2 * i + 1]) << 16);
#pragma unroll
    for (int i = 0; i < 8; ++i)
      lB[buf][(bn + i) * B_STR + bk] = f2h(b_reg[i]);
  };

  AccF acc[4];
#pragma unroll
  for (int q = 0; q < 4; ++q)
#pragma unroll
    for (int r = 0; r < 8; ++r) acc[q].f[r] = 0.0f;

  const int nk = K / GT_K;
  load_tile(0);
  store_tile(0);
  __syncthreads();

  for (int kt = 0; kt < nk; ++kt) {
    const int cur = kt & 1;
    if (kt + 1 < nk) load_tile((kt + 1) * GT_K);

    FragH fa0, fa1, fb0, fb1;
    const int am  = wm + fm;
    const int bnn = wn + fm;
#pragma unroll
    for (int j = 0; j < 8; ++j) {
      const int kba = (j < 4) ? (8 * fh + 2 * j) : (16 + 8 * fh + 2 * (j - 4));
      fa0.u[j] = *(const unsigned int*)&lA[cur][am * A_STR + kba];
      fa1.u[j] = *(const unsigned int*)&lA[cur][(am + 16) * A_STR + kba];
    }
#pragma unroll
    for (int j = 0; j < 8; ++j) {
      const int kbb = (j < 4) ? (8 * fh + 2 * j) : (16 + 8 * fh + 2 * (j - 4));
      fb0.u[j] = *(const unsigned int*)&lB[cur][bnn * B_STR + kbb];
      fb1.u[j] = *(const unsigned int*)&lB[cur][(bnn + 16) * B_STR + kbb];
    }
    acc[0].v = __builtin_amdgcn_wmma_f32_16x16x32_f16(false, fa0.v, false, fb0.v, (short)0, acc[0].v, false, false);
    acc[1].v = __builtin_amdgcn_wmma_f32_16x16x32_f16(false, fa0.v, false, fb1.v, (short)0, acc[1].v, false, false);
    acc[2].v = __builtin_amdgcn_wmma_f32_16x16x32_f16(false, fa1.v, false, fb0.v, (short)0, acc[2].v, false, false);
    acc[3].v = __builtin_amdgcn_wmma_f32_16x16x32_f16(false, fa1.v, false, fb1.v, (short)0, acc[3].v, false, false);

    if (kt + 1 < nk) store_tile(1 - cur);
    __syncthreads();
  }

  // ---- epilogue: bias + activation + store ----
#pragma unroll
  for (int ti = 0; ti < 2; ++ti) {
#pragma unroll
    for (int tj = 0; tj < 2; ++tj) {
      AccF& a = acc[ti * 2 + tj];
      const int gn = n0 + wn + tj * 16 + fm;
      const float bv = bias ? bias[gn] : 0.0f;
#pragma unroll
      for (int r = 0; r < 8; ++r) {
        const int gm = m0 + wm + ti * 16 + r + 8 * fh;
        if (gm < M) {
          float v = a.f[r] + bv;
          if (ACT == 1) v = v > 0.0f ? v : 0.0f;
          if (ACT == 2) v = tanhf(v);
          C[(size_t)gm * N + gn] = v;
        }
      }
    }
  }
}

// ---------------------------------------------------------------------------
// Elementwise / reduction kernels
// ---------------------------------------------------------------------------
static __device__ __forceinline__ float wave_sum(float v) {
#pragma unroll
  for (int off = 16; off > 0; off >>= 1) v += __shfl_xor(v, off, 32);
  return v;
}

// layernorm over rows of D; optional residual add and optional scalar scale
__global__ __launch_bounds__(256) void ln_rows(
    float* __restrict__ dst, const float* __restrict__ src,
    const float* __restrict__ addsrc, const float* __restrict__ g,
    const float* __restrict__ b, int rows, int D,
    const float* __restrict__ scaleptr)
{
  const int row  = blockIdx.x * 8 + (threadIdx.x >> 5);
  const int lane = threadIdx.x & 31;
  if (row >= rows) return;
  const float* s = src + (size_t)row * D;
  const float* a = addsrc ? addsrc + (size_t)row * D : nullptr;
  float sum = 0.0f, sq = 0.0f;
  for (int d = lane; d < D; d += 32) {
    float x = s[d] + (a ? a[d] : 0.0f);
    sum += x; sq += x * x;
  }
  sum = wave_sum(sum); sq = wave_sum(sq);
  const float mean = sum / (float)D;
  const float var  = sq / (float)D - mean * mean;
  const float inv  = rsqrtf(var + 1e-5f);
  const float sc   = scaleptr ? *scaleptr : 1.0f;
  float* o = dst + (size_t)row * D;
  for (int d = lane; d < D; d += 32) {
    float x = s[d] + (a ? a[d] : 0.0f);
    o[d] = ((x - mean) * inv * g[d] + b[d]) * sc;
  }
}

__global__ __launch_bounds__(256) void add_pe_k(float* __restrict__ X, int Mtok) {
  const size_t i = (size_t)blockIdx.x * blockDim.x + threadIdx.x;
  if (i >= (size_t)Mtok * HIDD) return;
  const int d   = (int)(i & (HIDD - 1));
  const int tok = (int)(i >> 9);
  const int s   = tok & (SEQL - 1);
  const int pr  = d >> 1;
  const float dv  = __expf((float)(2 * pr) * (-9.210340371976184f / (float)HIDD));
  const float ang = (float)s * dv;
  X[i] += (d & 1) ? __cosf(ang) : __sinf(ang);
}

__global__ __launch_bounds__(256) void attn_scores_k(
    const float* __restrict__ Q, const float* __restrict__ Kv,
    float* __restrict__ ATT, int Nseq)
{
  const int idx = blockIdx.x * blockDim.x + threadIdx.x;
  if (idx >= Nseq * NHEADS * SEQL) return;
  const int q   = idx & 7;
  const int h   = (idx >> 3) & 3;
  const int seq = idx >> 5;
  const float* qp = Q + (size_t)(seq * SEQL + q) * HIDD + h * HEADD;
  float s[SEQL];
  float mx = -1e30f;
#pragma unroll
  for (int kk = 0; kk < SEQL; ++kk) {
    const float* kp = Kv + (size_t)(seq * SEQL + kk) * HIDD + h * HEADD;
    float acc = 0.0f;
    for (int d = 0; d < HEADD; ++d) acc += qp[d] * kp[d];
    s[kk] = acc * 0.08838834764831845f;   // 1/sqrt(128)
    mx = fmaxf(mx, s[kk]);
  }
  float sum = 0.0f;
#pragma unroll
  for (int kk = 0; kk < SEQL; ++kk) { s[kk] = __expf(s[kk] - mx); sum += s[kk]; }
  const float inv = 1.0f / sum;
  float* op = ATT + (size_t)idx * SEQL;
#pragma unroll
  for (int kk = 0; kk < SEQL; ++kk) op[kk] = s[kk] * inv;
}

__global__ __launch_bounds__(256) void attn_apply_k(
    const float* __restrict__ ATT, const float* __restrict__ V,
    float* __restrict__ O, int Nseq)
{
  const int idx = blockIdx.x * blockDim.x + threadIdx.x;
  if (idx >= Nseq * NHEADS * SEQL * SEQL) return;
  const int dg  = idx & 7;
  const int q   = (idx >> 3) & 7;
  const int h   = (idx >> 6) & 3;
  const int seq = idx >> 8;
  const float* ap = ATT + (size_t)(seq * 32 + h * 8 + q) * SEQL;
  float acc[16];
#pragma unroll
  for (int i = 0; i < 16; ++i) acc[i] = 0.0f;
#pragma unroll
  for (int kk = 0; kk < SEQL; ++kk) {
    const float wgt = ap[kk];
    const float* vp = V + (size_t)(seq * SEQL + kk) * HIDD + h * HEADD + dg * 16;
#pragma unroll
    for (int i = 0; i < 16; ++i) acc[i] += wgt * vp[i];
  }
  float* op = O + (size_t)(seq * SEQL + q) * HIDD + h * HEADD + dg * 16;
#pragma unroll
  for (int i = 0; i < 16; ++i) op[i] = acc[i];
}

__global__ __launch_bounds__(256) void pool_logits_k(
    const float* __restrict__ X, const float* __restrict__ pw,
    float* __restrict__ logit, int Mtok)
{
  const int tok = blockIdx.x * blockDim.x + threadIdx.x;
  if (tok >= Mtok) return;
  const float* x = X + (size_t)tok * HIDD;
  float acc = 0.0f;
  for (int d = 0; d < HIDD; ++d) acc += x[d] * pw[d];
  logit[tok] = acc;
}

__global__ __launch_bounds__(256) void pool_softmax_k(
    const float* __restrict__ logit, float* __restrict__ pwgt, int Nseq)
{
  const int seq = blockIdx.x * blockDim.x + threadIdx.x;
  if (seq >= Nseq) return;
  float s[SEQL];
  float mx = -1e30f;
#pragma unroll
  for (int i = 0; i < SEQL; ++i) { s[i] = logit[seq * SEQL + i]; mx = fmaxf(mx, s[i]); }
  float sum = 0.0f;
#pragma unroll
  for (int i = 0; i < SEQL; ++i) { s[i] = __expf(s[i] - mx); sum += s[i]; }
  const float inv = 1.0f / sum;
#pragma unroll
  for (int i = 0; i < SEQL; ++i) pwgt[seq * SEQL + i] = s[i] * inv;
}

__global__ __launch_bounds__(256) void pool_apply_k(
    const float* __restrict__ pwgt, const float* __restrict__ X,
    float* __restrict__ pooled, int Nseq)
{
  const int idx = blockIdx.x * blockDim.x + threadIdx.x;
  if (idx >= Nseq * HIDD) return;
  const int d   = idx & (HIDD - 1);
  const int seq = idx >> 9;
  float acc = 0.0f;
#pragma unroll
  for (int s = 0; s < SEQL; ++s)
    acc += pwgt[seq * SEQL + s] * X[(size_t)(seq * SEQL + s) * HIDD + d];
  pooled[idx] = acc;
}

__global__ __launch_bounds__(256) void gelu_k(float* __restrict__ x, size_t n) {
  const size_t i = (size_t)blockIdx.x * blockDim.x + threadIdx.x;
  if (i >= n) return;
  const float v = x[i];
  x[i] = 0.5f * v * (1.0f + erff(v * 0.7071067811865476f));
}

__global__ __launch_bounds__(256) void fuse_combine_k(
    const float* __restrict__ a, const float* __restrict__ b,
    const float* __restrict__ twp, const float* __restrict__ swp,
    float* __restrict__ o, size_t n)
{
  const size_t i = (size_t)blockIdx.x * blockDim.x + threadIdx.x;
  if (i >= n) return;
  const float tw = 1.0f / (1.0f + __expf(-*twp));
  const float sw = 1.0f / (1.0f + __expf(-*swp));
  o[i] = tw * a[i] + sw * b[i];
}

__global__ __launch_bounds__(256) void spmm_scatter_k(
    const float* __restrict__ vals, const int* __restrict__ rows,
    const int* __restrict__ cols, const float* __restrict__ ego,
    float* __restrict__ side, int nnz)
{
  const int idx = blockIdx.x * blockDim.x + threadIdx.x;
  if (idx >= nnz * 8) return;
  const int e  = idx >> 3;
  const int d0 = (idx & 7) * 16;
  const float v = vals[e];
  const int r = rows[e], c = cols[e];
  const float* src = ego + (size_t)c * EMB + d0;
  float* dst = side + (size_t)r * EMB + d0;
#pragma unroll
  for (int i = 0; i < 16; ++i) atomicAdd(&dst[i], v * src[i]);
}

// sigmoid( concat(a_row, b_row) @ W[256] + bias )
__global__ __launch_bounds__(256) void gate_sig_k(
    const float* __restrict__ A, const float* __restrict__ B,
    const float* __restrict__ W, const float* __restrict__ bptr,
    float* __restrict__ out, int rows)
{
  const int r = blockIdx.x * blockDim.x + threadIdx.x;
  if (r >= rows) return;
  const float* a = A + (size_t)r * EMB;
  const float* b = B + (size_t)r * EMB;
  float acc = *bptr;
  for (int d = 0; d < EMB; ++d) acc += a[d] * W[d] + b[d] * W[EMB + d];
  out[r] = 1.0f / (1.0f + __expf(-acc));
}

__global__ __launch_bounds__(256) void gate_combine_k(
    const float* __restrict__ f, const float* __restrict__ g,
    const float* __restrict__ s, const float* __restrict__ c,
    float* __restrict__ o, size_t n)
{
  const size_t i = (size_t)blockIdx.x * blockDim.x + threadIdx.x;
  if (i >= n) return;
  const size_t r = i >> 7;
  o[i] = f[r] * s[i] + g[r] * c[i];
}

__global__ __launch_bounds__(256) void out_gate_k(
    const float* __restrict__ cell, const float* __restrict__ Wo,
    const float* __restrict__ bo, float* __restrict__ outp, int rows)
{
  const int r = blockIdx.x * blockDim.x + threadIdx.x;
  if (r >= rows) return;
  const float* c = cell + (size_t)r * EMB;
  float acc = *bo;
  for (int d = 0; d < EMB; ++d) acc += c[d] * Wo[d];
  const float o = 1.0f / (1.0f + __expf(-acc));
  float* op = outp + (size_t)r * EMB;
  for (int d = 0; d < EMB; ++d) op[d] = o * c[d];
}

__global__ __launch_bounds__(256) void final_combine_k(
    const float* __restrict__ e0, const float* __restrict__ l0,
    const float* __restrict__ l1, const float* __restrict__ l2,
    float* __restrict__ fin, size_t n)
{
  const size_t i = (size_t)blockIdx.x * blockDim.x + threadIdx.x;
  if (i >= n) return;
  fin[i] = 0.2f * e0[i] + (0.8f / 3.0f) * (l0[i] + l1[i] + l2[i]);
}

__global__ __launch_bounds__(256) void score_out_k(
    const float* __restrict__ fin, const int* __restrict__ users,
    const int* __restrict__ items, float* __restrict__ out, int batch)
{
  const int b = blockIdx.x * blockDim.x + threadIdx.x;
  if (b >= batch) return;
  const float* up = fin + (size_t)users[b] * EMB;
  const float* ip = fin + (size_t)(N_USER + items[b]) * EMB;
  float acc = 0.0f;
  for (int d = 0; d < EMB; ++d) acc += up[d] * ip[d];
  out[b] = acc;
}

// ---------------------------------------------------------------------------
// Host orchestration
// ---------------------------------------------------------------------------
static inline int cdiv(long long a, long long b) { return (int)((a + b - 1) / b); }

static void launch_gemm(int act, const float* A, const float* B, const float* bias,
                        float* C, int M, int K, int N, hipStream_t s) {
  dim3 grid(N / GT_N, cdiv(M, GT_M));
  if (act == 0)      gemm_wmma_f16<0><<<grid, 256, 0, s>>>(A, B, bias, C, M, K, N);
  else if (act == 1) gemm_wmma_f16<1><<<grid, 256, 0, s>>>(A, B, bias, C, M, K, N);
  else               gemm_wmma_f16<2><<<grid, 256, 0, s>>>(A, B, bias, C, M, K, N);
}

static void launch_ln(float* dst, const float* src, const float* add,
                      const float* g, const float* b, int rows, int D,
                      const float* scale, hipStream_t s) {
  ln_rows<<<cdiv(rows, 8), 256, 0, s>>>(dst, src, add, g, b, rows, D, scale);
}

static void run_transformer(const float* const* F, int tf, const float* seq_in, int Nseq,
                            float* bX, float* bY, float* bQ, float* bK, float* bV,
                            float* bFF, float* bATT, float* bLOG, float* bPW,
                            float* bPOOL, float* tout, hipStream_t s)
{
  const int Mtok = Nseq * SEQL;
  // input LN (D=64) into bFF scratch
  launch_ln(bFF, seq_in, nullptr, F[tf + TF_IN_LN_G], F[tf + TF_IN_LN_B], Mtok, FEAT, nullptr, s);
  // proj 64->512
  launch_gemm(0, bFF, F[tf + TF_PROJ_W], F[tf + TF_PROJ_B], bX, Mtok, FEAT, HIDD, s);
  add_pe_k<<<cdiv((long long)Mtok * HIDD, 256), 256, 0, s>>>(bX, Mtok);
  for (int l = 0; l < 2; ++l) {
    const int lb = tf + (l == 0 ? TF_LAYER0 : TF_LAYER1);
    launch_gemm(0, bX, F[lb + L_WQ], F[lb + L_BQ], bQ, Mtok, HIDD, HIDD, s);
    launch_gemm(0, bX, F[lb + L_WK], F[lb + L_BK], bK, Mtok, HIDD, HIDD, s);
    launch_gemm(0, bX, F[lb + L_WV], F[lb + L_BV], bV, Mtok, HIDD, HIDD, s);
    attn_scores_k<<<cdiv(Nseq * NHEADS * SEQL, 256), 256, 0, s>>>(bQ, bK, bATT, Nseq);
    attn_apply_k<<<cdiv(Nseq * NHEADS * SEQL * SEQL, 256), 256, 0, s>>>(bATT, bV, bY, Nseq);
    launch_gemm(0, bY, F[lb + L_WO], F[lb + L_BO], bQ, Mtok, HIDD, HIDD, s);
    launch_ln(bX, bX, bQ, F[lb + L_LN1_G], F[lb + L_LN1_B], Mtok, HIDD, nullptr, s);
    launch_gemm(1, bX, F[lb + L_FF1_W], F[lb + L_FF1_B], bFF, Mtok, HIDD, 2 * HIDD, s);
    launch_gemm(0, bFF, F[lb + L_FF2_W], F[lb + L_FF2_B], bY, Mtok, 2 * HIDD, HIDD, s);
    launch_ln(bX, bX, bY, F[lb + L_LN2_G], F[lb + L_LN2_B], Mtok, HIDD, nullptr, s);
  }
  pool_logits_k<<<cdiv(Mtok, 256), 256, 0, s>>>(bX, F[tf + TF_POOL_W], bLOG, Mtok);
  pool_softmax_k<<<cdiv(Nseq, 256), 256, 0, s>>>(bLOG, bPW, Nseq);
  pool_apply_k<<<cdiv(Nseq * HIDD, 256), 256, 0, s>>>(bPW, bX, bPOOL, Nseq);
  launch_gemm(1, bPOOL, F[tf + TF_FC_MID_W], F[tf + TF_FC_MID_B], bQ, Nseq, HIDD, HIDD, s);
  launch_ln(bQ, bQ, nullptr, F[tf + TF_LN_G], F[tf + TF_LN_B], Nseq, HIDD, nullptr, s);
  launch_gemm(0, bQ, F[tf + TF_FC_OUT_W], F[tf + TF_FC_OUT_B], tout, Nseq, HIDD, EMB, s);
  launch_ln(tout, tout, nullptr, F[tf + TF_OUT_LN_G], F[tf + TF_OUT_LN_B], Nseq, EMB, nullptr, s);
}

static void run_fusion(const float* const* F, int fu, const float* t, const float* st,
                       int rows, float* tmpA, float* tmpB, float* outp, hipStream_t s)
{
  const size_t n = (size_t)rows * EMB;
  launch_gemm(0, t, F[fu + FU_T_W], F[fu + FU_T_B], tmpA, rows, EMB, EMB, s);
  launch_ln(tmpA, tmpA, nullptr, F[fu + FU_T_LN_G], F[fu + FU_T_LN_B], rows, EMB, nullptr, s);
  gelu_k<<<cdiv(n, 256), 256, 0, s>>>(tmpA, n);
  launch_gemm(0, st, F[fu + FU_S_W], F[fu + FU_S_B], tmpB, rows, STRUCT_D, EMB, s);
  launch_ln(tmpB, tmpB, nullptr, F[fu + FU_S_LN_G], F[fu + FU_S_LN_B], rows, EMB, nullptr, s);
  gelu_k<<<cdiv(n, 256), 256, 0, s>>>(tmpB, n);
  fuse_combine_k<<<cdiv(n, 256), 256, 0, s>>>(tmpA, tmpB, F[fu + FU_TW], F[fu + FU_SW], tmpA, n);
  launch_gemm(0, tmpA, F[fu + FU_F_W], F[fu + FU_F_B], outp, rows, EMB, EMB, s);
  launch_ln(outp, outp, nullptr, F[fu + FU_F_LN_G], F[fu + FU_F_LN_B], rows, EMB, nullptr, s);
  gelu_k<<<cdiv(n, 256), 256, 0, s>>>(outp, n);
}

extern "C" void kernel_launch(void* const* d_in, const int* in_sizes, int n_in,
                              void* d_out, int out_size, void* d_ws, size_t ws_size,
                              hipStream_t stream)
{
  (void)in_sizes; (void)n_in; (void)out_size; (void)ws_size;
  const float* const* F = (const float* const*)d_in;
  float* ws = (float*)d_ws;

  // ---- workspace carve-out (floats) ----
  size_t off = 0;
  auto carve = [&](size_t nfloats) { float* p = ws + off; off += nfloats; return p; };
  float* bX    = carve((size_t)MAXTOK * HIDD);
  float* bY    = carve((size_t)MAXTOK * HIDD);
  float* bQ    = carve((size_t)MAXTOK * HIDD);
  float* bK    = carve((size_t)MAXTOK * HIDD);
  float* bV    = carve((size_t)MAXTOK * HIDD);
  float* bFF   = carve((size_t)MAXTOK * 2 * HIDD);
  float* bATT  = carve((size_t)N_ITEM * NHEADS * SEQL * SEQL);
  float* bLOG  = carve((size_t)MAXTOK);
  float* bPW   = carve((size_t)MAXTOK);
  float* bPOOL = carve((size_t)N_ITEM * HIDD);
  float* bTC   = carve((size_t)N_ITEM * EMB);   // circ transformer out
  float* bTM   = carve((size_t)N_USER * EMB);   // mirna transformer out
  float* bFA   = carve((size_t)NN * EMB);
  float* bFB   = carve((size_t)NN * EMB);
  float* bEGO0 = carve((size_t)NN * EMB);
  float* bSIDE = carve((size_t)NN * EMB);
  float* bSELF = carve((size_t)NN * EMB);
  float* bCROSS= carve((size_t)NN * EMB);
  float* bTMP  = carve((size_t)NN * EMB);
  float* bCELL = carve((size_t)NN * EMB);
  float* bNEW  = carve((size_t)NN * EMB);
  float* bFg   = carve((size_t)NN);
  float* bIg   = carve((size_t)NN);
  float* bL0   = carve((size_t)NN * EMB);
  float* bL1   = carve((size_t)NN * EMB);
  float* bL2   = carve((size_t)NN * EMB);
  float* bFIN  = carve((size_t)NN * EMB);

  // ---- transformers ----
  run_transformer(F, IDX_CIRC_TF, F[IDX_CIRC_SEQ], N_ITEM,
                  bX, bY, bQ, bK, bV, bFF, bATT, bLOG, bPW, bPOOL, bTC, stream);
  run_transformer(F, IDX_MIR_TF, F[IDX_MIR_SEQ], N_USER,
                  bX, bY, bQ, bK, bV, bFF, bATT, bLOG, bPW, bPOOL, bTM, stream);

  // ---- fusion: ego0 = concat(mirna_e[0..NU), circ_e[NU..NN)) ----
  run_fusion(F, IDX_MIR_FUS, bTM, F[IDX_MIR_STRUCT], N_USER, bFA, bFB, bEGO0, stream);
  run_fusion(F, IDX_CIRC_FUS, bTC, F[IDX_CIRC_STRUCT], N_ITEM, bFA, bFB,
             bEGO0 + (size_t)N_USER * EMB, stream);

  // ---- graph propagation ----
  const float* ego = bEGO0;
  float* louts[3] = {bL0, bL1, bL2};
  const float* adj_vals = F[IDX_ADJ_VALS];
  const int* adj_row = (const int*)d_in[IDX_ADJ_ROW];
  const int* adj_col = (const int*)d_in[IDX_ADJ_COL];
  for (int k = 0; k < 3; ++k) {
    const int pb = IDX_PROP + k * 27;
    hipMemsetAsync(bSIDE, 0, (size_t)NN * EMB * sizeof(float), stream);
    spmm_scatter_k<<<cdiv((long long)NNZ_E * 8, 256), 256, 0, stream>>>(
        adj_vals, adj_row, adj_col, ego, bSIDE, NNZ_E);
    // self/cross transforms (users then items)
    launch_gemm(0, ego, F[pb + P_W_M], F[pb + P_B_M], bSELF, N_USER, EMB, EMB, stream);
    launch_gemm(0, bSIDE, F[pb + P_W_C_TO_M], F[pb + P_B_C_TO_M], bCROSS, N_USER, EMB, EMB, stream);
    launch_gemm(0, ego + (size_t)N_USER * EMB, F[pb + P_W_C], F[pb + P_B_C],
                bSELF + (size_t)N_USER * EMB, N_ITEM, EMB, EMB, stream);
    launch_gemm(0, bSIDE + (size_t)N_USER * EMB, F[pb + P_W_M_TO_C], F[pb + P_B_M_TO_C],
                bCROSS + (size_t)N_USER * EMB, N_ITEM, EMB, EMB, stream);
    // gates
    gate_sig_k<<<cdiv(N_USER, 256), 256, 0, stream>>>(
        ego, bSIDE, F[pb + P_W_F_M], F[pb + P_B_F_M], bFg, N_USER);
    gate_sig_k<<<cdiv(N_ITEM, 256), 256, 0, stream>>>(
        ego + (size_t)N_USER * EMB, bSIDE + (size_t)N_USER * EMB,
        F[pb + P_W_F_C], F[pb + P_B_F_C], bFg + N_USER, N_ITEM);
    gate_sig_k<<<cdiv(N_USER, 256), 256, 0, stream>>>(
        bSELF, bCROSS, F[pb + P_W_I_M], F[pb + P_B_I_M], bIg, N_USER);
    gate_sig_k<<<cdiv(N_ITEM, 256), 256, 0, stream>>>(
        bSELF + (size_t)N_USER * EMB, bCROSS + (size_t)N_USER * EMB,
        F[pb + P_W_I_C], F[pb + P_B_I_C], bIg + N_USER, N_ITEM);
    gate_combine_k<<<cdiv((long long)NN * EMB, 256), 256, 0, stream>>>(
        bFg, bIg, bSELF, bCROSS, bTMP, (size_t)NN * EMB);
    // cell = tanh(tmp @ W_ec + b_ec)   (fused tanh in GEMM epilogue)
    launch_gemm(2, bTMP, F[pb + P_W_EC_M], F[pb + P_B_EC_M], bCELL, N_USER, EMB, EMB, stream);
    launch_gemm(2, bTMP + (size_t)N_USER * EMB, F[pb + P_W_EC_C], F[pb + P_B_EC_C],
                bCELL + (size_t)N_USER * EMB, N_ITEM, EMB, EMB, stream);
    // out = sigmoid(cell @ W_o + b_o) * cell
    out_gate_k<<<cdiv(N_USER, 256), 256, 0, stream>>>(
        bCELL, F[pb + P_W_O_M], F[pb + P_B_O_M], bNEW, N_USER);
    out_gate_k<<<cdiv(N_ITEM, 256), 256, 0, stream>>>(
        bCELL + (size_t)N_USER * EMB, F[pb + P_W_O_C], F[pb + P_B_O_C],
        bNEW + (size_t)N_USER * EMB, N_ITEM);
    // layernorm * scale -> layer output (pointer-swap as next ego)
    launch_ln(louts[k], bNEW, nullptr, F[pb + P_LN_G], F[pb + P_LN_B],
              NN, EMB, F[pb + P_SCALE], stream);
    ego = louts[k];
  }

  // ---- final combine + scoring ----
  final_combine_k<<<cdiv((long long)NN * EMB, 256), 256, 0, stream>>>(
      bEGO0, bL0, bL1, bL2, bFIN, (size_t)NN * EMB);
  score_out_k<<<cdiv(BATCH_B, 256), 256, 0, stream>>>(
      bFIN, (const int*)d_in[IDX_USERS], (const int*)d_in[IDX_ITEMS],
      (float*)d_out, BATCH_B);
}
